// GCN1_21792664060289
// MI455X (gfx1250) — matrix operand order repaired and checked by reading, not compiled, not run on hardware
//
#include <hip/hip_runtime.h>
#include <hip/hip_bf16.h>
#include <math.h>

// ---------------------------------------------------------------- constants
#define BATCH   128
#define NEDGE   512
#define IMGSZ   256
#define PATCH   32
#define CH      384
#define HEADS   3
#define DEPTH   6
#define NPATCH  64
#define NTOK    65          // 64 patches + cls
#define NPAD    80          // padded to 5*16 for WMMA tiles
#define HD      128
#define MLP_H   1536
#define ROWS    (BATCH * NTOK)      // 8320 = 520*16
#define EROWS   (NEDGE * NTOK)      // 33280 = 2080*16
#define SCALE_ATTN 0.08838834764831845f  // 128^-0.5

typedef __attribute__((ext_vector_type(16))) __bf16 v16bf;
typedef __attribute__((ext_vector_type(8)))  __bf16 v8bf;
typedef __attribute__((ext_vector_type(8)))  float  v8f;

// ---------------------------------------------------------------- helpers
static __device__ __forceinline__ unsigned short f2bfu(float f) {
  unsigned u = __builtin_bit_cast(unsigned, f);
  unsigned r = u + 0x7FFFu + ((u >> 16) & 1u);   // round-to-nearest-even
  return (unsigned short)(r >> 16);
}
static __device__ __forceinline__ float bfu2f(unsigned short s) {
  unsigned u = ((unsigned)s) << 16;
  return __builtin_bit_cast(float, u);
}
// fragment: 16 bf16 = two aligned 16-byte chunks
static __device__ __forceinline__ v16bf load_frag16(const unsigned short* p0,
                                                    const unsigned short* p1) {
  v8bf a = *reinterpret_cast<const v8bf*>(p0);
  v8bf b = *reinterpret_cast<const v8bf*>(p1);
  return __builtin_shufflevector(a, b, 0,1,2,3,4,5,6,7,8,9,10,11,12,13,14,15);
}

// ---------------------------------------------------------------- fp32 -> bf16 bulk convert (weights)
__global__ __launch_bounds__(256)
void cvt_bf16_kernel(const float* __restrict__ src, unsigned short* __restrict__ dst, int n) {
  int i = blockIdx.x * 256 + threadIdx.x;
  if (i < n) dst[i] = f2bfu(src[i]);
}

// ---------------------------------------------------------------- edge lists (deterministic scatter)
__global__ void edges_kernel(const int* __restrict__ edge, int* __restrict__ elist,
                             int* __restrict__ ecount, float* __restrict__ cnt) {
  int b = threadIdx.x;
  if (b >= BATCH) return;
  int c = 0;
  for (int e = 0; e < NEDGE; ++e)
    if (edge[NEDGE + e] == b) elist[b * NEDGE + (c++)] = e;   // dst row
  ecount[b] = c;
  cnt[b] = (c > 0) ? (float)c : 1.0f;
}

// ---------------------------------------------------------------- im2col (patch gather -> bf16)
__global__ __launch_bounds__(256)
void im2col_kernel(const float* __restrict__ img, unsigned short* __restrict__ A) {
  size_t idx = (size_t)blockIdx.x * 256 + threadIdx.x;
  if (idx >= (size_t)BATCH * NPATCH * 3072) return;
  int k = (int)(idx % 3072);
  int m = (int)(idx / 3072);
  int b = m >> 6, p = m & 63;
  int ci = k >> 10, r = k & 1023, kh = r >> 5, kw = r & 31;
  int py = p >> 3, px = p & 7;
  float v = img[(((size_t)b * 3 + ci) * IMGSZ + py * PATCH + kh) * IMGSZ + px * PATCH + kw];
  A[idx] = f2bfu(v);
}

// ---------------------------------------------------------------- bf16 WMMA GEMM, 16x64 strip / wave
// out[row][col] = sum_k A[row][k] * W[col][k]  (+bias) (+gelu) ; A,W bf16, W row-major [N][K]
template <bool GELU_EPI, bool REMAP, bool ACCUM>
__global__ __launch_bounds__(256)
void gemm_bf16_kernel(const unsigned short* __restrict__ A, const unsigned short* __restrict__ Wb,
                      const float* __restrict__ bias,
                      float* __restrict__ outF, unsigned short* __restrict__ outB,
                      int N, int K, int tilesNG, int numTiles) {
  int wave = threadIdx.x >> 5;
  int tile = blockIdx.x * 8 + wave;
  if (tile >= numTiles) return;
  int tm = tile / tilesNG, tg = tile % tilesNG;
  int lane = threadIdx.x & 31;
  int l = lane & 15, hi = lane >> 4;
  const unsigned short* ap  = A + (size_t)(tm * 16 + l) * K;
  const unsigned short* wp0 = Wb + (size_t)(tg * 64 + 0 * 16 + l) * K + 16 * hi;
  const unsigned short* wp1 = Wb + (size_t)(tg * 64 + 1 * 16 + l) * K + 16 * hi;
  const unsigned short* wp2 = Wb + (size_t)(tg * 64 + 2 * 16 + l) * K + 16 * hi;
  const unsigned short* wp3 = Wb + (size_t)(tg * 64 + 3 * 16 + l) * K + 16 * hi;
  v8f acc0 = {0.f,0.f,0.f,0.f,0.f,0.f,0.f,0.f};
  v8f acc1 = acc0, acc2 = acc0, acc3 = acc0;
  for (int k0 = 0; k0 < K; k0 += 32) {
    v16bf af = load_frag16(ap + k0 + 16 * hi, ap + k0 + 16 * hi + 8);
    v16bf b0 = load_frag16(wp0 + k0, wp0 + k0 + 8);
    v16bf b1 = load_frag16(wp1 + k0, wp1 + k0 + 8);
    v16bf b2 = load_frag16(wp2 + k0, wp2 + k0 + 8);
    v16bf b3 = load_frag16(wp3 + k0, wp3 + k0 + 8);
    acc0 = __builtin_amdgcn_wmma_f32_16x16x32_bf16(false, af, false, b0, (short)0, acc0, false, false);
    acc1 = __builtin_amdgcn_wmma_f32_16x16x32_bf16(false, af, false, b1, (short)0, acc1, false, false);
    acc2 = __builtin_amdgcn_wmma_f32_16x16x32_bf16(false, af, false, b2, (short)0, acc2, false, false);
    acc3 = __builtin_amdgcn_wmma_f32_16x16x32_bf16(false, af, false, b3, (short)0, acc3, false, false);
  }
#pragma unroll
  for (int j = 0; j < 4; ++j) {
    v8f acc = (j == 0) ? acc0 : (j == 1) ? acc1 : (j == 2) ? acc2 : acc3;
    int col = tg * 64 + j * 16 + l;
    float bb = bias ? bias[col] : 0.f;
#pragma unroll
    for (int r = 0; r < 8; ++r) {
      int row = tm * 16 + r + 8 * hi;
      float v = acc[r] + bb;
      if (GELU_EPI) v = 0.5f * v * (1.0f + erff(v * 0.70710678118654752f));
      size_t orow = row;
      if (REMAP) orow = (size_t)(row / NPATCH) * NTOK + (row % NPATCH) + 1;  // patch -> token slot
      size_t oi = orow * (size_t)N + col;
      if (outF) { if (ACCUM) outF[oi] += v; else outF[oi] = v; }
      if (outB) outB[oi] = f2bfu(v);
    }
  }
}

// ---------------------------------------------------------------- x = cls/pos assembly
__global__ __launch_bounds__(256)
void assemble_kernel(float* __restrict__ x, const float* __restrict__ cls,
                     const float* __restrict__ pos) {
  int idx = blockIdx.x * 256 + threadIdx.x;
  if (idx >= ROWS * CH) return;
  int c = idx % CH;
  int n = (idx / CH) % NTOK;
  if (n == 0) x[idx] = cls[c] + pos[c];
  else        x[idx] += pos[n * CH + c];
}

// ---------------------------------------------------------------- LayerNorm (wave per row) -> bf16
__global__ __launch_bounds__(256)
void ln_bf16_kernel(const float* __restrict__ x, const float* __restrict__ w,
                    const float* __restrict__ b, unsigned short* __restrict__ out, int rows) {
  int wave = threadIdx.x >> 5, lane = threadIdx.x & 31;
  int row = blockIdx.x * 8 + wave;
  if (row >= rows) return;
  const float* xr = x + (size_t)row * CH;
  float v[12], s = 0.f;
#pragma unroll
  for (int i = 0; i < 12; ++i) { v[i] = xr[lane + i * 32]; s += v[i]; }
  for (int off = 16; off; off >>= 1) s += __shfl_xor(s, off, 32);
  float mu = s * (1.0f / CH);
  float vs = 0.f;
#pragma unroll
  for (int i = 0; i < 12; ++i) { float d = v[i] - mu; vs += d * d; }
  for (int off = 16; off; off >>= 1) vs += __shfl_xor(vs, off, 32);
  float inv = rsqrtf(vs * (1.0f / CH) + 1e-5f);
#pragma unroll
  for (int i = 0; i < 12; ++i) {
    int c = lane + i * 32;
    out[(size_t)row * CH + c] = f2bfu((v[i] - mu) * inv * w[c] + b[c]);
  }
}

// final LN on cls token only -> fp32 output [B, C]
__global__ __launch_bounds__(256)
void ln_final_kernel(const float* __restrict__ x, const float* __restrict__ w,
                     const float* __restrict__ b, float* __restrict__ out) {
  int wave = threadIdx.x >> 5, lane = threadIdx.x & 31;
  int bb = blockIdx.x * 8 + wave;
  if (bb >= BATCH) return;
  const float* xr = x + (size_t)bb * NTOK * CH;   // token 0
  float v[12], s = 0.f;
#pragma unroll
  for (int i = 0; i < 12; ++i) { v[i] = xr[lane + i * 32]; s += v[i]; }
  for (int off = 16; off; off >>= 1) s += __shfl_xor(s, off, 32);
  float mu = s * (1.0f / CH);
  float vs = 0.f;
#pragma unroll
  for (int i = 0; i < 12; ++i) { float d = v[i] - mu; vs += d * d; }
  for (int off = 16; off; off >>= 1) vs += __shfl_xor(vs, off, 32);
  float inv = rsqrtf(vs * (1.0f / CH) + 1e-5f);
#pragma unroll
  for (int i = 0; i < 12; ++i) {
    int c = lane + i * 32;
    out[(size_t)bb * CH + c] = (v[i] - mu) * inv * w[c] + b[c];
  }
}

// ---------------------------------------------------------------- per-(edge,head) attention
// Q,V from dst node; K from src node. Tokens padded 65 -> 80, K-dim of PV padded 80 -> 96.
__global__ __launch_bounds__(256)
void attn_kernel(const unsigned short* __restrict__ qkvb, const int* __restrict__ edge,
                 unsigned short* __restrict__ o) {
  extern __shared__ char smem[];
  unsigned short* Qs  = (unsigned short*)smem;                 // [80][128] bf16
  unsigned short* Ks  = Qs + NPAD * HD;                        // [80][128]
  unsigned short* VsT = Ks + NPAD * HD;                        // [128][96]  (hd-major)
  float*          Ss  = (float*)(smem + 65536);                // [80][80] fp32
  unsigned short* Ps  = (unsigned short*)(smem + 65536 + NPAD * NPAD * 4); // [80][96]

  int e = blockIdx.x / HEADS, hh = blockIdx.x % HEADS;
  int srcb = edge[e], dstb = edge[NEDGE + e];
  int tid = threadIdx.x;

  for (int i = tid; i < NPAD * HD; i += 256) {
    int n = i >> 7, d = i & 127;
    unsigned short q = 0, k = 0;
    if (n < NTOK) {
      size_t qb = ((size_t)(dstb * NTOK + n)) * (3 * CH);
      size_t kb = ((size_t)(srcb * NTOK + n)) * (3 * CH);
      q = qkvb[qb + 0 * CH + hh * HD + d];
      k = qkvb[kb + 1 * CH + hh * HD + d];
    }
    Qs[i] = q; Ks[i] = k;
  }
  for (int i = tid; i < HD * 96; i += 256) {
    int d = i / 96, m = i % 96;
    unsigned short v = 0;
    if (m < NTOK)
      v = qkvb[((size_t)(dstb * NTOK + m)) * (3 * CH) + 2 * CH + hh * HD + d];
    VsT[i] = v;
  }
  __syncthreads();

  int wave = tid >> 5, lane = tid & 31, l = lane & 15, hi = lane >> 4;

  // ---- S = (Q K^T) * scale : 5x5 tiles, K-dim 128
  for (int t = wave; t < 25; t += 8) {
    int ti = t / 5, tj = t % 5;
    v8f acc = {0.f,0.f,0.f,0.f,0.f,0.f,0.f,0.f};
    const unsigned short* ap = Qs + (size_t)(ti * 16 + l) * HD;
    const unsigned short* bp = Ks + (size_t)(tj * 16 + l) * HD;
#pragma unroll
    for (int kk = 0; kk < 4; ++kk) {
      int k0 = kk * 32;
      v16bf af  = load_frag16(ap + k0 + 16 * hi, ap + k0 + 16 * hi + 8);
      v16bf bfv = load_frag16(bp + k0 + 16 * hi, bp + k0 + 16 * hi + 8);
      acc = __builtin_amdgcn_wmma_f32_16x16x32_bf16(false, af, false, bfv,
                                                    (short)0, acc, false, false);
    }
#pragma unroll
    for (int r = 0; r < 8; ++r)
      Ss[(ti * 16 + r + 8 * hi) * NPAD + tj * 16 + l] = acc[r] * SCALE_ATTN;
  }
  __syncthreads();

  // ---- masked softmax, wave-parallel rows (valid keys 0..64) -> bf16 P [80][96]
  for (int r = wave; r < NPAD; r += 8) {
    const float* srow = Ss + r * NPAD;
    unsigned short* prow = Ps + r * 96;
    float sv[3];
#pragma unroll
    for (int i = 0; i < 3; ++i) {
      int col = lane + 32 * i;
      sv[i] = (col < NTOK) ? srow[col] : -1e30f;
    }
    float m = fmaxf(sv[0], fmaxf(sv[1], sv[2]));
    for (int off = 16; off; off >>= 1) m = fmaxf(m, __shfl_xor(m, off, 32));
    float p[3], s = 0.f;
#pragma unroll
    for (int i = 0; i < 3; ++i) {
      int col = lane + 32 * i;
      p[i] = (col < NTOK) ? expf(sv[i] - m) : 0.f;
      s += p[i];
    }
    for (int off = 16; off; off >>= 1) s += __shfl_xor(s, off, 32);
    float inv = 1.0f / s;
#pragma unroll
    for (int i = 0; i < 3; ++i) {
      int col = lane + 32 * i;
      prow[col] = f2bfu(p[i] * inv);
    }
  }
  __syncthreads();

  // ---- O = P V : 5x8 tiles, K-dim 96 (zero padded)
  for (int t = wave; t < 40; t += 8) {
    int ti = t / 8, tn = t % 8;
    v8f acc = {0.f,0.f,0.f,0.f,0.f,0.f,0.f,0.f};
    const unsigned short* ap = Ps  + (size_t)(ti * 16 + l) * 96;
    const unsigned short* bp = VsT + (size_t)(tn * 16 + l) * 96;
#pragma unroll
    for (int kk = 0; kk < 3; ++kk) {
      int k0 = kk * 32;
      v16bf af  = load_frag16(ap + k0 + 16 * hi, ap + k0 + 16 * hi + 8);
      v16bf bfv = load_frag16(bp + k0 + 16 * hi, bp + k0 + 16 * hi + 8);
      acc = __builtin_amdgcn_wmma_f32_16x16x32_bf16(false, af, false, bfv,
                                                    (short)0, acc, false, false);
    }
#pragma unroll
    for (int r = 0; r < 8; ++r) {
      int row = ti * 16 + r + 8 * hi;
      if (row < NTOK)
        o[((size_t)(e * NTOK + row)) * CH + hh * HD + tn * 16 + l] = f2bfu(acc[r]);
    }
  }
}

// ---------------------------------------------------------------- deterministic scatter-mean + residual
__global__ __launch_bounds__(256)
void scatter_kernel(const unsigned short* __restrict__ msg, const int* __restrict__ elist,
                    const int* __restrict__ ecount, const float* __restrict__ cnt,
                    float* __restrict__ x) {
  int idx = blockIdx.x * 256 + threadIdx.x;
  if (idx >= ROWS * CH) return;
  int c = idx % CH;
  int row = idx / CH;
  int b = row / NTOK, n = row % NTOK;
  int ne = ecount[b];
  float s = 0.f;
  for (int j = 0; j < ne; ++j) {
    int e = elist[b * NEDGE + j];
    s += bfu2f(msg[((size_t)(e * NTOK + n)) * CH + c]);
  }
  x[idx] += s / cnt[b];
}

// ---------------------------------------------------------------- host
extern "C" void kernel_launch(void* const* d_in, const int* in_sizes, int n_in,
                              void* d_out, int out_size, void* d_ws, size_t ws_size,
                              hipStream_t stream) {
  const float* images  = (const float*)d_in[0];
  const float* patch_w = (const float*)d_in[1];
  const float* patch_b = (const float*)d_in[2];
  const float* cls_tok = (const float*)d_in[3];
  const float* pos_emb = (const float*)d_in[4];
  const float* norm1_w = (const float*)d_in[5];
  const float* norm1_b = (const float*)d_in[6];
  const float* qkv_w   = (const float*)d_in[7];
  const float* proj_w  = (const float*)d_in[8];
  const float* proj_b  = (const float*)d_in[9];
  const float* norm2_w = (const float*)d_in[10];
  const float* norm2_b = (const float*)d_in[11];
  const float* fc1_w   = (const float*)d_in[12];
  const float* fc1_b   = (const float*)d_in[13];
  const float* fc2_w   = (const float*)d_in[14];
  const float* fc2_b   = (const float*)d_in[15];
  const float* norm_w  = (const float*)d_in[16];
  const float* norm_b  = (const float*)d_in[17];
  const int*   edge    = (const int*)d_in[18];
  float* out = (float*)d_out;

  char* ws = (char*)d_ws;
  size_t off = 0;
  auto take = [&](size_t bytes) { size_t o = off; off = (off + bytes + 255) & ~(size_t)255; return o; };

  float*          x     = (float*)(ws + take((size_t)ROWS * CH * 4));
  unsigned short* xb    = (unsigned short*)(ws + take((size_t)ROWS * CH * 2));
  unsigned short* qkvb  = (unsigned short*)(ws + take((size_t)ROWS * 3 * CH * 2));
  unsigned short* hbuf  = (unsigned short*)(ws + take((size_t)ROWS * MLP_H * 2));
  // bf16 weight copies (converted once)
  const int NPW  = CH * 3072;
  const int NQW  = DEPTH * 3 * CH * CH;
  const int NPJ  = DEPTH * CH * CH;
  const int NF1  = DEPTH * MLP_H * CH;
  const int NF2  = DEPTH * CH * MLP_H;
  unsigned short* wpatch = (unsigned short*)(ws + take((size_t)NPW * 2));
  unsigned short* wqkv   = (unsigned short*)(ws + take((size_t)NQW * 2));
  unsigned short* wproj  = (unsigned short*)(ws + take((size_t)NPJ * 2));
  unsigned short* wfc1   = (unsigned short*)(ws + take((size_t)NF1 * 2));
  unsigned short* wfc2   = (unsigned short*)(ws + take((size_t)NF2 * 2));
  // shared region: im2col buffer (prologue only) overlaps o/msg (layer loop)
  size_t region = take((size_t)EROWS * CH * 2 * 2 > (size_t)BATCH * NPATCH * 3072 * 2
                           ? (size_t)EROWS * CH * 2 * 2
                           : (size_t)BATCH * NPATCH * 3072 * 2);
  unsigned short* Abuf = (unsigned short*)(ws + region);
  unsigned short* obuf = (unsigned short*)(ws + region);
  unsigned short* msg  = obuf + (size_t)EROWS * CH;
  int*   elist  = (int*)(ws + take((size_t)BATCH * NEDGE * 4));
  int*   ecount = (int*)(ws + take((size_t)BATCH * 4));
  float* cntf   = (float*)(ws + take((size_t)BATCH * 4));

  // prologue: edge lists + weight conversion + im2col
  edges_kernel<<<1, 128, 0, stream>>>(edge, elist, ecount, cntf);
  cvt_bf16_kernel<<<(NPW + 255) / 256, 256, 0, stream>>>(patch_w, wpatch, NPW);
  cvt_bf16_kernel<<<(NQW + 255) / 256, 256, 0, stream>>>(qkv_w,  wqkv,  NQW);
  cvt_bf16_kernel<<<(NPJ + 255) / 256, 256, 0, stream>>>(proj_w, wproj, NPJ);
  cvt_bf16_kernel<<<(NF1 + 255) / 256, 256, 0, stream>>>(fc1_w,  wfc1,  NF1);
  cvt_bf16_kernel<<<(NF2 + 255) / 256, 256, 0, stream>>>(fc2_w,  wfc2,  NF2);
  {
    size_t tot = (size_t)BATCH * NPATCH * 3072;
    im2col_kernel<<<(int)((tot + 255) / 256), 256, 0, stream>>>(images, Abuf);
  }
  // patch embed GEMM: [8192 x 3072] x [3072 x 384] -> x (token slots 1..64)
  {
    int tilesNG = CH / 64, nt = (BATCH * NPATCH / 16) * tilesNG;
    gemm_bf16_kernel<false, true, false><<<(nt + 7) / 8, 256, 0, stream>>>(
        Abuf, wpatch, patch_b, x, (unsigned short*)nullptr, CH, 3072, tilesNG, nt);
  }
  assemble_kernel<<<(ROWS * CH + 255) / 256, 256, 0, stream>>>(x, cls_tok, pos_emb);

  const size_t ATTN_LDS = 65536 + (size_t)NPAD * NPAD * 4 + (size_t)NPAD * 96 * 2; // 106496

  for (int d = 0; d < DEPTH; ++d) {
    // LN1 -> bf16
    ln_bf16_kernel<<<(ROWS + 7) / 8, 256, 0, stream>>>(x, norm1_w + d * CH, norm1_b + d * CH, xb, ROWS);
    // qkv GEMM (no bias) -> bf16 [ROWS][1152]
    {
      int tilesNG = (3 * CH) / 64, nt = (ROWS / 16) * tilesNG;
      gemm_bf16_kernel<false, false, false><<<(nt + 7) / 8, 256, 0, stream>>>(
          xb, wqkv + (size_t)d * 3 * CH * CH, nullptr, (float*)nullptr, qkvb, 3 * CH, CH, tilesNG, nt);
    }
    // per-edge attention -> o bf16 [E*65][384]
    attn_kernel<<<NEDGE * HEADS, 256, ATTN_LDS, stream>>>(qkvb, edge, obuf);
    // proj GEMM (+bias) -> msg bf16 [E*65][384]
    {
      int tilesNG = CH / 64, nt = (EROWS / 16) * tilesNG;
      gemm_bf16_kernel<false, false, false><<<(nt + 7) / 8, 256, 0, stream>>>(
          obuf, wproj + (size_t)d * CH * CH, proj_b + d * CH, (float*)nullptr, msg, CH, CH, tilesNG, nt);
    }
    // scatter-mean + residual into x
    scatter_kernel<<<(ROWS * CH + 255) / 256, 256, 0, stream>>>(msg, elist, ecount, cntf, x);
    // LN2 -> bf16
    ln_bf16_kernel<<<(ROWS + 7) / 8, 256, 0, stream>>>(x, norm2_w + d * CH, norm2_b + d * CH, xb, ROWS);
    // fc1 (+bias, exact GELU) -> h bf16
    {
      int tilesNG = MLP_H / 64, nt = (ROWS / 16) * tilesNG;
      gemm_bf16_kernel<true, false, false><<<(nt + 7) / 8, 256, 0, stream>>>(
          xb, wfc1 + (size_t)d * MLP_H * CH, fc1_b + (size_t)d * MLP_H,
          (float*)nullptr, hbuf, MLP_H, CH, tilesNG, nt);
    }
    // fc2 (+bias) accumulated into x (residual)
    {
      int tilesNG = CH / 64, nt = (ROWS / 16) * tilesNG;
      gemm_bf16_kernel<false, false, true><<<(nt + 7) / 8, 256, 0, stream>>>(
          hbuf, wfc2 + (size_t)d * CH * MLP_H, fc2_b + d * CH, x, (unsigned short*)nullptr,
          CH, MLP_H, tilesNG, nt);
    }
  }

  // final LN on cls token
  ln_final_kernel<<<(BATCH + 7) / 8, 256, 0, stream>>>(x, norm_w, norm_b, out);
  (void)in_sizes; (void)n_in; (void)out_size; (void)ws_size;
}
